// SanaMSAdaLNBlock_36567351558922
// MI455X (gfx1250) — matrix operand order repaired and checked by reading, not compiled, not run on hardware
//
#include <hip/hip_runtime.h>

typedef _Float16 half_t;
typedef __attribute__((ext_vector_type(16))) _Float16 v16h;
typedef __attribute__((ext_vector_type(8)))  _Float16 v8h;
typedef __attribute__((ext_vector_type(8)))  float    v8f;

#define B_     4
#define C_     288
#define PIX_   4096      // 64*64
#define D_     1152
#define HF_    4608
#define HEADS_ 36
#define S_     1026      // 1024 patches + 2 prefix tokens
#define NP_    1024
#define C6_    1728

__device__ __forceinline__ float silu_f(float x) { return x / (1.f + __expf(-x)); }

// ---------------------------------------------------------------------------
// Generic WMMA GEMM: C[M,N] = act(A[M,K] * B[K,N] + bias[N])
// A,B fp16 row-major; C fp32 or fp16. Block tile 64x64, K step 32, 8 waves.
// A tile staged via CDNA5 async global->LDS DMA (ASYNCcnt), B tile staged
// transposed via registers. Fragments read with ds_load_b128, math in
// v_wmma_f32_16x16x32_f16.
// ---------------------------------------------------------------------------
#define BM 64
#define BN 64
#define BK 32
#define LDA_S 40   // LDS row stride (halves), padded
#define LDB_S 40

template<bool OUT_F16, bool ACT_SILU, bool HAS_BIAS>
__global__ __launch_bounds__(256) void k_gemm(
    const half_t* __restrict__ A, long long strideA, int lda,
    const half_t* __restrict__ Bm, long long strideB, int ldb,
    void* __restrict__ Cm, long long strideC, int ldc,
    const float* __restrict__ bias, int M, int N, int K)
{
  __shared__ half_t sA[BM * LDA_S];
  __shared__ half_t sB[BN * LDB_S];   // transposed: [n][k]

  const int bz = blockIdx.z;
  A  += (long long)bz * strideA;
  Bm += (long long)bz * strideB;

  const int tid  = threadIdx.x;
  const int wave = tid >> 5;
  const int lane = tid & 31;
  const int m0 = blockIdx.y * BM;
  const int n0 = blockIdx.x * BN;

  const int wm = (wave & 3) * 16;   // 0,16,32,48
  const int wn = (wave >> 2) * 32;  // 0 or 32
  const int lmod  = lane & 15;
  const int lhalf = (lane < 16) ? 0 : 1;

  // global->LDS load mapping
  const int a_row  = tid >> 2;        // 0..63
  const int a_koff = (tid & 3) * 8;   // 0,8,16,24
  const int b_krow = tid >> 3;        // 0..31
  const int b_noff = (tid & 7) * 8;   // 0..56

  // Rows >= M contribute only to C rows >= M (discarded at store), so the
  // A address may be clamped instead of zero-filled.
  int gm = m0 + a_row; if (gm >= M) gm = M - 1;
  const half_t* aptr = A + (long long)gm * lda + a_koff;           // advance by BK
  const half_t* bptr = Bm + (long long)b_krow * ldb + n0 + b_noff; // advance by BK*ldb
  const unsigned ldsA = (unsigned)(unsigned long long)(const void*)
                        &sA[a_row * LDA_S + a_koff];

  v8f acc0 = {}; v8f acc1 = {};

  for (int k0 = 0; k0 < K; k0 += BK) {
    // stage A tile [64][32]: per-lane 16B async DMA, global -> LDS
    asm volatile("global_load_async_to_lds_b128 %0, %1, off"
                 :: "v"(ldsA), "v"((unsigned long long)(size_t)aptr)
                 : "memory");
    // stage B tile, transposed to [n][k]
    {
      v8h bv = *(const v8h*)bptr;
      #pragma unroll
      for (int j = 0; j < 8; ++j)
        sB[(b_noff + j) * LDB_S + b_krow] = bv[j];
    }
    if (k0 + BK < K) {
      __builtin_prefetch(aptr + BK, 0, 3);
      __builtin_prefetch(bptr + (long long)BK * ldb, 0, 3);
    }
    aptr += BK;
    bptr += (long long)BK * ldb;
    asm volatile("s_wait_asynccnt 0x0" ::: "memory");
    __syncthreads();

    // A fragment: lane<16 -> K 0..15, lane>=16 -> K 16..31
    const half_t* pa = &sA[(wm + lmod) * LDA_S + lhalf * 16];
    v8h alo = *(const v8h*)pa;
    v8h ahi = *(const v8h*)(pa + 8);
    v16h afrag = __builtin_shufflevector(alo, ahi,
        0,1,2,3,4,5,6,7,8,9,10,11,12,13,14,15);

    // B fragments: lane<16 -> K 0..15, lane>=16 -> K 16..31 (contig in sB row)
    const half_t* pb0 = &sB[(wn + lmod) * LDB_S + lhalf * 16];
    v8h b0lo = *(const v8h*)pb0;
    v8h b0hi = *(const v8h*)(pb0 + 8);
    v16h bfrag0 = __builtin_shufflevector(b0lo, b0hi,
        0,1,2,3,4,5,6,7,8,9,10,11,12,13,14,15);
    const half_t* pb1 = pb0 + 16 * LDB_S;
    v8h b1lo = *(const v8h*)pb1;
    v8h b1hi = *(const v8h*)(pb1 + 8);
    v16h bfrag1 = __builtin_shufflevector(b1lo, b1hi,
        0,1,2,3,4,5,6,7,8,9,10,11,12,13,14,15);

    acc0 = __builtin_amdgcn_wmma_f32_16x16x32_f16(false, afrag, false, bfrag0,
                                                  (short)0, acc0, false, false);
    acc1 = __builtin_amdgcn_wmma_f32_16x16x32_f16(false, afrag, false, bfrag1,
                                                  (short)0, acc1, false, false);
    __syncthreads();
  }

  // epilogue
  const int col0 = n0 + wn + lmod;
  float bias0 = 0.f, bias1 = 0.f;
  if (HAS_BIAS) { bias0 = bias[col0]; bias1 = bias[col0 + 16]; }
  #pragma unroll
  for (int r = 0; r < 8; ++r) {
    int row = m0 + wm + r + lhalf * 8;
    if (row < M) {
      float v0 = acc0[r] + bias0;
      float v1 = acc1[r] + bias1;
      if (ACT_SILU) { v0 = silu_f(v0); v1 = silu_f(v1); }
      if (OUT_F16) {
        half_t* C = (half_t*)Cm + (long long)bz * strideC;
        C[(long long)row * ldc + col0]      = (half_t)v0;
        C[(long long)row * ldc + col0 + 16] = (half_t)v1;
      } else {
        float* C = (float*)Cm + (long long)bz * strideC;
        C[(long long)row * ldc + col0]      = v0;
        C[(long long)row * ldc + col0 + 16] = v1;
      }
    }
  }
}

// ---------------------------------------------------------------------------
// Elementwise / conversion kernels
// ---------------------------------------------------------------------------
__global__ __launch_bounds__(256) void k_silu_f16(const float* __restrict__ in,
                                                  half_t* __restrict__ out, long long n) {
  long long i = (long long)blockIdx.x * 256 + threadIdx.x;
  if (i < n) out[i] = (half_t)silu_f(in[i]);
}

__global__ __launch_bounds__(256) void k_cvt_f16(const float* __restrict__ in,
                                                 half_t* __restrict__ out, long long n) {
  long long i = (long long)blockIdx.x * 256 + threadIdx.x;
  if (i < n) out[i] = (half_t)in[i];
}

// out[c][r] = (f16) in[r][c] ; R rows, Ccols cols
__global__ __launch_bounds__(256) void k_transpose_f16(const float* __restrict__ in,
                                                       half_t* __restrict__ out,
                                                       int R, int Ccols) {
  __shared__ float tile[32][33];
  int r0 = blockIdx.y * 32, c0 = blockIdx.x * 32;
  int tx = threadIdx.x, ty = threadIdx.y;   // 32 x 8
  #pragma unroll
  for (int j = 0; j < 32; j += 8) {
    int r = r0 + ty + j, c = c0 + tx;
    tile[ty + j][tx] = (r < R && c < Ccols) ? in[(long long)r * Ccols + c] : 0.f;
  }
  __syncthreads();
  #pragma unroll
  for (int j = 0; j < 32; j += 8) {
    int c = c0 + ty + j, r = r0 + tx;
    if (c < Ccols && r < R) out[(long long)c * R + r] = (half_t)tile[tx][ty + j];
  }
}

// ---------------------------------------------------------------------------
// LayerNorm reductions (global per-batch mean/var over C*H*W)
// ---------------------------------------------------------------------------
__global__ void k_zero_stats(double* stats) {
  if (threadIdx.x < 16) stats[threadIdx.x] = 0.0;
}

__global__ __launch_bounds__(256) void k_ln_partial(const float* __restrict__ xin,
                                                    double* stats, long long perBatch) {
  int b = blockIdx.y;
  const float* p = xin + (long long)b * perBatch;
  long long base = (long long)blockIdx.x * 4096 + threadIdx.x;
  float s = 0.f, q = 0.f;
  #pragma unroll
  for (int j = 0; j < 16; ++j) {
    long long idx = base + j * 256;
    if (idx < perBatch) { float v = p[idx]; s += v; q += v * v; }
  }
  __shared__ float rs_[256], rq_[256];
  rs_[threadIdx.x] = s; rq_[threadIdx.x] = q;
  __syncthreads();
  for (int st = 128; st > 0; st >>= 1) {
    if (threadIdx.x < st) {
      rs_[threadIdx.x] += rs_[threadIdx.x + st];
      rq_[threadIdx.x] += rq_[threadIdx.x + st];
    }
    __syncthreads();
  }
  if (threadIdx.x == 0) {
    atomicAdd(&stats[b * 2 + 0], (double)rs_[0]);
    atomicAdd(&stats[b * 2 + 1], (double)rq_[0]);
  }
}

__global__ void k_ln_final(const double* __restrict__ stats, float* __restrict__ muRs) {
  int b = threadIdx.x;
  if (b < B_) {
    double n = (double)C_ * PIX_;
    double mu = stats[b * 2] / n;
    double var = stats[b * 2 + 1] / n - mu * mu;
    muRs[b * 2]     = (float)mu;
    muRs[b * 2 + 1] = (float)(1.0 / sqrt(var + 1e-6));
  }
}

// ---------------------------------------------------------------------------
// Token build: tok[row, cc] = f16( (x-mu)*rs * (1+sc) + sh ), segmented
// ---------------------------------------------------------------------------
__global__ __launch_bounds__(256) void k_build_tok(
    const float* __restrict__ x, const float* __restrict__ ss,
    const float* __restrict__ ss_b, const float* __restrict__ muRs,
    half_t* __restrict__ out, int chSh, int chSc, int rowStride, int rowOff)
{
  long long i = (long long)blockIdx.x * 256 + threadIdx.x;   // B*C*PIX
  int pix = (int)(i % PIX_);
  int c   = (int)((i / PIX_) % C_);
  int b   = (int)(i / ((long long)PIX_ * C_));
  int h = pix >> 6, w = pix & 63;
  float mu = muRs[b * 2], rs = muRs[b * 2 + 1];
  float val = (x[i] - mu) * rs;
  float sc = ss[((long long)b * C6_ + chSc * C_ + c) * PIX_ + pix] + ss_b[chSc * C_ + c];
  float sh = ss[((long long)b * C6_ + chSh * C_ + c) * PIX_ + pix] + ss_b[chSh * C_ + c];
  float t = val * (1.f + sc) + sh;
  int n  = (h >> 1) * 32 + (w >> 1);
  int cc = c * 4 + (h & 1) * 2 + (w & 1);
  out[((long long)(b * rowStride + rowOff + n)) * D_ + cc] = (half_t)t;
}

__global__ __launch_bounds__(256) void k_prefix(const float* __restrict__ g,
                                                const float* __restrict__ t,
                                                half_t* __restrict__ tok) {
  int i = blockIdx.x * 256 + threadIdx.x;
  if (i >= B_ * 2 * D_) return;
  int c = i % D_;
  int r = (i / D_) & 1;
  int b = i / (2 * D_);
  const float* src = r ? t : g;
  tok[((long long)b * S_ + r) * D_ + c] = (half_t)src[b * D_ + c];
}

// ---------------------------------------------------------------------------
// Linear attention: vk[bh][e][d] = sum_s vpad[s,e] * relu(k[s,d]) ; e=32 -> v=1
// ---------------------------------------------------------------------------
__global__ __launch_bounds__(256) void k_attn_vk(const float* __restrict__ qkv,
                                                 float* __restrict__ vk) {
  int bh = blockIdx.x;                 // b*36+h
  int b = bh / HEADS_, hh = bh % HEADS_;
  __shared__ float sk[8][32], sv[8][32];
  float acc[5] = {0.f, 0.f, 0.f, 0.f, 0.f};
  int tid = threadIdx.x;

  for (int s0 = 0; s0 < S_; s0 += 8) {
    #pragma unroll
    for (int it = 0; it < 2; ++it) {
      int idx = tid + it * 256;        // 0..511
      int sr  = idx >> 6;              // row in chunk
      int col = idx & 63;              // <32: k, else v
      int s = s0 + sr;
      float val = 0.f;
      if (s < S_) {
        long long row = (long long)(b * S_ + s) * (3 * D_);
        if (col < 32) val = fmaxf(qkv[row + D_ + hh * 32 + col], 0.f);
        else          val = qkv[row + 2 * D_ + hh * 32 + (col - 32)];
      }
      if (col < 32) sk[sr][col] = val; else sv[sr][col - 32] = val;
    }
    __syncthreads();
    #pragma unroll
    for (int o = 0; o < 4; ++o) {
      int oid = tid + o * 256;         // 0..1023
      int e = oid >> 5, d = oid & 31;
      float a = acc[o];
      #pragma unroll
      for (int sr = 0; sr < 8; ++sr) a += sv[sr][e] * sk[sr][d];
      acc[o] = a;
    }
    if (tid < 32) {                     // pad row e=32 (v==1)
      float a = acc[4];
      #pragma unroll
      for (int sr = 0; sr < 8; ++sr) a += sk[sr][tid];
      acc[4] = a;
    }
    __syncthreads();
  }
  long long base = (long long)bh * 33 * 32;
  #pragma unroll
  for (int o = 0; o < 4; ++o) vk[base + tid + o * 256] = acc[o];
  if (tid < 32) vk[base + 1024 + tid] = acc[4];
}

// out[s, h*32+e] = (sum_d vk[e,d]*relu(q_d)) / (sum_d vk[32,d]*relu(q_d) + eps)
__global__ __launch_bounds__(256) void k_attn_out(const float* __restrict__ qkv,
                                                  const float* __restrict__ vk,
                                                  half_t* __restrict__ out) {
  int gw = blockIdx.x * 8 + (threadIdx.x >> 5);
  int lane = threadIdx.x & 31;
  int total = B_ * S_ * HEADS_;
  if (gw >= total) return;
  int hh = gw % HEADS_;
  int s  = (gw / HEADS_) % S_;
  int b  = gw / (HEADS_ * S_);
  long long qrow = (long long)(b * S_ + s) * (3 * D_);
  float q = fmaxf(qkv[qrow + hh * 32 + lane], 0.f);
  long long vkb = (long long)(b * HEADS_ + hh) * 33 * 32;
  float r[32];
  #pragma unroll
  for (int d = 0; d < 32; ++d) r[d] = vk[vkb + lane * 32 + d];
  float num = 0.f;
  #pragma unroll
  for (int d = 0; d < 32; ++d) num += r[d] * __shfl(q, d, 32);
  float dp = vk[vkb + 1024 + lane] * q;
  #pragma unroll
  for (int off = 16; off > 0; off >>= 1) dp += __shfl_xor(dp, off, 32);
  float o = num / (dp + 1e-8f);
  out[(long long)(b * S_ + s) * D_ + hh * 32 + lane] = (half_t)o;
}

// ---------------------------------------------------------------------------
// Combine: out = base + comb(tok) * (ss[gchunk] + ss_b)
// ---------------------------------------------------------------------------
__global__ __launch_bounds__(256) void k_combine(
    const float* __restrict__ base, const float* __restrict__ tokbuf,
    const float* __restrict__ ss, const float* __restrict__ ss_b,
    int chG, int rowStride, int rowOff, float* __restrict__ out)
{
  long long i = (long long)blockIdx.x * 256 + threadIdx.x;   // B*C*PIX
  int pix = (int)(i % PIX_);
  int c   = (int)((i / PIX_) % C_);
  int b   = (int)(i / ((long long)PIX_ * C_));
  int h = pix >> 6, w = pix & 63;
  int n  = (h >> 1) * 32 + (w >> 1);
  int cc = c * 4 + (h & 1) * 2 + (w & 1);
  float g = ss[((long long)b * C6_ + chG * C_ + c) * PIX_ + pix] + ss_b[chG * C_ + c];
  float t = tokbuf[(long long)(b * rowStride + rowOff + n) * D_ + cc];
  out[i] = base[i] + t * g;
}

// ---------------------------------------------------------------------------
// Depthwise 3x3 + GLU gate, token-major fp16 in/out
// ---------------------------------------------------------------------------
__global__ __launch_bounds__(256) void k_dwglu(const half_t* __restrict__ xin,
                                               const float* __restrict__ dw_w,
                                               const float* __restrict__ dw_b,
                                               half_t* __restrict__ out) {
  long long i = (long long)blockIdx.x * 256 + threadIdx.x;   // B*1024*4608
  int c = (int)(i % HF_);
  long long bp = i / HF_;
  int pix = (int)(bp % NP_);
  int b = (int)(bp / NP_);
  int hp = pix >> 5, wp = pix & 31;
  float a1 = dw_b[c], a2 = dw_b[c + HF_];
  #pragma unroll
  for (int ky = 0; ky < 3; ++ky) {
    int hh = hp + ky - 1;
    if ((unsigned)hh >= 32u) continue;
    #pragma unroll
    for (int kx = 0; kx < 3; ++kx) {
      int ww = wp + kx - 1;
      if ((unsigned)ww >= 32u) continue;
      long long rowb = ((long long)b * NP_ + hh * 32 + ww) * (2 * HF_);
      a1 += (float)xin[rowb + c]       * dw_w[c * 9 + ky * 3 + kx];
      a2 += (float)xin[rowb + HF_ + c] * dw_w[(c + HF_) * 9 + ky * 3 + kx];
    }
  }
  out[bp * HF_ + c] = (half_t)(a1 * silu_f(a2));
}

// ---------------------------------------------------------------------------
extern "C" void kernel_launch(void* const* d_in, const int* in_sizes, int n_in,
                              void* d_out, int out_size, void* d_ws, size_t ws_size,
                              hipStream_t stream) {
  (void)in_sizes; (void)n_in; (void)out_size; (void)ws_size;
  const float* x      = (const float*)d_in[0];
  const float* cond   = (const float*)d_in[1];
  const float* gtok   = (const float*)d_in[2];
  const float* ttok   = (const float*)d_in[3];
  const float* ss_w   = (const float*)d_in[4];
  const float* ss_b   = (const float*)d_in[5];
  const float* qkv_w  = (const float*)d_in[6];
  const float* proj_w = (const float*)d_in[7];
  const float* proj_b = (const float*)d_in[8];
  const float* inv_w  = (const float*)d_in[9];
  const float* inv_b  = (const float*)d_in[10];
  const float* dw_w   = (const float*)d_in[11];
  const float* dw_b   = (const float*)d_in[12];
  const float* pw_w   = (const float*)d_in[13];
  float* out = (float*)d_out;

  char* ws = (char*)d_ws;
  size_t off = 0;
  auto alloc = [&](size_t bytes) {
    size_t o = off; off += (bytes + 255) & ~(size_t)255; return o;
  };
  half_t* condH = (half_t*)(ws + alloc((size_t)B_ * C6_ * PIX_ * 2));
  half_t* ssW   = (half_t*)(ws + alloc((size_t)C6_ * C6_ * 2));
  half_t* qkvW  = (half_t*)(ws + alloc((size_t)D_ * 3 * D_ * 2));
  half_t* projW = (half_t*)(ws + alloc((size_t)D_ * D_ * 2));
  half_t* invW  = (half_t*)(ws + alloc((size_t)D_ * 2 * HF_ * 2));
  half_t* pwW   = (half_t*)(ws + alloc((size_t)HF_ * D_ * 2));
  float*  ssO   = (float*) (ws + alloc((size_t)B_ * C6_ * PIX_ * 4));
  half_t* tok   = (half_t*)(ws + alloc((size_t)B_ * S_ * D_ * 2));
  float*  qkvO  = (float*) (ws + alloc((size_t)B_ * S_ * 3 * D_ * 4));
  float*  vk    = (float*) (ws + alloc((size_t)B_ * HEADS_ * 33 * 32 * 4));
  half_t* attnH = (half_t*)(ws + alloc((size_t)B_ * S_ * D_ * 2));
  float*  projO = (float*) (ws + alloc((size_t)B_ * S_ * D_ * 4));
  float*  x2    = (float*) (ws + alloc((size_t)B_ * C_ * PIX_ * 4));
  half_t* t2    = (half_t*)(ws + alloc((size_t)B_ * NP_ * D_ * 2));
  half_t* invO  = (half_t*)(ws + alloc((size_t)B_ * NP_ * 2 * HF_ * 2));
  half_t* glu   = (half_t*)(ws + alloc((size_t)B_ * NP_ * HF_ * 2));
  float*  pwO   = (float*) (ws + alloc((size_t)B_ * NP_ * D_ * 4));
  double* stats = (double*)(ws + alloc(32 * 8));
  float*  muRs  = (float*) (ws + alloc(32 * 4));

  // ---- prep: zero stats, fp16 conversions / weight transposes
  k_zero_stats<<<1, 32, 0, stream>>>(stats);
  k_silu_f16<<<110592, 256, 0, stream>>>(cond, condH, (long long)B_ * C6_ * PIX_);
  k_cvt_f16<<<11664, 256, 0, stream>>>(ss_w, ssW, (long long)C6_ * C6_);
  dim3 tb(32, 8);
  k_transpose_f16<<<dim3(D_ / 32, 3 * D_ / 32), tb, 0, stream>>>(qkv_w, qkvW, 3 * D_, D_);
  k_transpose_f16<<<dim3(D_ / 32, D_ / 32),     tb, 0, stream>>>(proj_w, projW, D_, D_);
  k_transpose_f16<<<dim3(D_ / 32, 2 * HF_ / 32),tb, 0, stream>>>(inv_w, invW, 2 * HF_, D_);
  k_transpose_f16<<<dim3(HF_ / 32, D_ / 32),    tb, 0, stream>>>(pw_w, pwW, D_, HF_);

  // ---- scale-shift GEMM: ss[b][o][pix] = ss_w x silu(cond)   (M=1728,N=4096,K=1728)
  k_gemm<false, false, false><<<dim3(PIX_ / 64, C6_ / 64, B_), 256, 0, stream>>>(
      ssW, 0, C6_, condH, (long long)C6_ * PIX_, PIX_,
      ssO, (long long)C6_ * PIX_, PIX_, nullptr, C6_, PIX_, C6_);

  // ---- LayerNorm #1 on x
  k_ln_partial<<<dim3(288, B_), 256, 0, stream>>>(x, stats, (long long)C_ * PIX_);
  k_ln_final<<<1, 32, 0, stream>>>(stats, muRs);

  // ---- tokens (prefix + modulated patches)
  k_prefix<<<36, 256, 0, stream>>>(gtok, ttok, tok);
  k_build_tok<<<18432, 256, 0, stream>>>(x, ssO, ss_b, muRs, tok, 0, 1, S_, 2);

  // ---- QKV GEMM (M=4104,N=3456,K=1152)
  k_gemm<false, false, false><<<dim3(3 * D_ / 64, 65, 1), 256, 0, stream>>>(
      tok, 0, D_, qkvW, 0, 3 * D_,
      qkvO, 0, 3 * D_, nullptr, B_ * S_, 3 * D_, D_);

  // ---- linear attention
  k_attn_vk<<<B_ * HEADS_, 256, 0, stream>>>(qkvO, vk);
  k_attn_out<<<18468, 256, 0, stream>>>(qkvO, vk, attnH);

  // ---- proj GEMM (+bias) (M=4104,N=1152,K=1152)
  k_gemm<false, false, true><<<dim3(D_ / 64, 65, 1), 256, 0, stream>>>(
      attnH, 0, D_, projW, 0, D_,
      projO, 0, D_, proj_b, B_ * S_, D_, D_);

  // ---- residual + gate -> x2
  k_combine<<<18432, 256, 0, stream>>>(x, projO, ssO, ss_b, 2, S_, 2, x2);

  // ---- LayerNorm #2 on x2
  k_ln_partial<<<dim3(288, B_), 256, 0, stream>>>(x2, stats + 8, (long long)C_ * PIX_);
  k_ln_final<<<1, 32, 0, stream>>>(stats + 8, muRs + 8);

  // ---- MLP tokens
  k_build_tok<<<18432, 256, 0, stream>>>(x2, ssO, ss_b, muRs + 8, t2, 3, 4, NP_, 0);

  // ---- inverted-bottleneck GEMM + bias + SiLU, fp16 out (M=4096,N=9216,K=1152)
  k_gemm<true, true, true><<<dim3(2 * HF_ / 64, NP_ * B_ / 64, 1), 256, 0, stream>>>(
      t2, 0, D_, invW, 0, 2 * HF_,
      invO, 0, 2 * HF_, inv_b, B_ * NP_, 2 * HF_, D_);

  // ---- depthwise 3x3 + GLU
  k_dwglu<<<73728, 256, 0, stream>>>(invO, dw_w, dw_b, glu);

  // ---- pointwise GEMM (M=4096,N=1152,K=4608)
  k_gemm<false, false, false><<<dim3(D_ / 64, NP_ * B_ / 64, 1), 256, 0, stream>>>(
      glu, 0, HF_, pwW, 0, D_,
      pwO, 0, D_, nullptr, B_ * NP_, D_, HF_);

  // ---- final residual + gate -> out
  k_combine<<<18432, 256, 0, stream>>>(x2, pwO, ssO, ss_b, 5, NP_, 0, out);
}
